// InteractionBlock_34797825032835
// MI455X (gfx1250) — matrix operand order repaired and checked by reading, not compiled, not run on hardware
//
#include <hip/hip_runtime.h>
#include <math.h>

#define INV_SQRT2f 0.70710678118654752440f
#define INV_SQRT3f 0.57735026918962576451f

typedef __bf16 bf16;
typedef __attribute__((ext_vector_type(16))) __bf16 v16bf;
typedef __attribute__((ext_vector_type(8)))  float  v8f;

union BQ { v16bf v; uint4 q[2]; };

__device__ __forceinline__ float siluf(float v) { return v / (1.0f + __expf(-v)); }

// ============================================================================
// WMMA GEMM (large-N): out[M,N] = epi( A[M,K](bf16) @ Bt[N,K](bf16)^T )
// Block = 128 threads (4 waves, 2x2 wave grid) -> 128x128 block tile.
// Each wave: 64x64 macro-tile = 16 v_wmma_f32_16x16x32_bf16 accumulators.
// Per K-step: 4 A-frags + 4 B-frags (16 b128 loads) feed 16 WMMAs
// (0.5 fragment-loads per WMMA; loads batch, WMMAs burst).
// A fragment (16-bit 16x32, ISA 7.12.2): lanes 0-15 rows M+0..15 K chunks
// {+0..7, +16..23}; lanes 16-31 same rows, K chunks {+8..15, +24..31}.
// B fragment: Bt row (= column of B): same K chunks as A per lane,
// lanes 0-15 K {+0..7, +16..23}, lanes 16-31 K {+8..15, +24..31}.
// K must be a multiple of 32.
// ============================================================================
__global__ __launch_bounds__(128)
void gemm_bf16_big_kernel(const bf16* __restrict__ A, const bf16* __restrict__ Bt,
                          float* __restrict__ Cf, bf16* __restrict__ Cb,
                          const float* __restrict__ Res,
                          int M, int N, int K, int act, float alpha)
{
    const int lane = threadIdx.x & 31;
    const int wave = threadIdx.x >> 5;
    const int hlf  = lane >> 4;
    const int l15  = lane & 15;
    const int row0  = blockIdx.x * 128 + (wave >> 1) * 64;
    const int nbase = blockIdx.y * 128 + (wave & 1) * 64;

    const bf16* Ap[4];
    const bf16* Bp[4];
#pragma unroll
    for (int s = 0; s < 4; ++s) {
        int ar = row0 + s * 16 + l15; if (ar > M - 1) ar = M - 1;
        Ap[s] = A + (size_t)ar * K;
        int bc = nbase + s * 16 + l15; if (bc > N - 1) bc = N - 1;
        Bp[s] = Bt + (size_t)bc * K;
    }

    v8f acc[4][4] = {};
    for (int k = 0; k < K; k += 32) {
        BQ af[4], bq[4];
#pragma unroll
        for (int s = 0; s < 4; ++s) {
            af[s].q[0] = *(const uint4*)(Ap[s] + k + hlf * 8);
            af[s].q[1] = *(const uint4*)(Ap[s] + k + 16 + hlf * 8);
        }
#pragma unroll
        for (int t = 0; t < 4; ++t) {
            bq[t].q[0] = *(const uint4*)(Bp[t] + k + hlf * 8);
            bq[t].q[1] = *(const uint4*)(Bp[t] + k + 16 + hlf * 8);
        }
        if (k + 32 < K) {
            __builtin_prefetch(Ap[0] + k + 32, 0, 1);
            __builtin_prefetch(Bp[0] + k + 32, 0, 1);
        }
#pragma unroll
        for (int s = 0; s < 4; ++s)
#pragma unroll
            for (int t = 0; t < 4; ++t)
                acc[s][t] = __builtin_amdgcn_wmma_f32_16x16x32_bf16(
                    false, af[s].v, false, bq[t].v, (short)0, acc[s][t], false, false);
    }

#pragma unroll
    for (int s = 0; s < 4; ++s) {
#pragma unroll
        for (int t = 0; t < 4; ++t) {
            int col = nbase + t * 16 + l15;
            if (col >= N) continue;
#pragma unroll
            for (int r = 0; r < 8; ++r) {
                int row = row0 + s * 16 + hlf * 8 + r;
                if (row >= M) continue;
                float v = acc[s][t][r];
                if (act) v = siluf(v);
                if (Res) v = (v + Res[(size_t)row * N + col]) * alpha;
                else     v = v * alpha;
                if (Cf) Cf[(size_t)row * N + col] = v;
                if (Cb) Cb[(size_t)row * N + col] = (bf16)v;
            }
        }
    }
}

// ============================================================================
// WMMA GEMM (small-N, N <= 64): 64x64 block tile, each wave 16 rows x 64 cols.
// ============================================================================
__global__ __launch_bounds__(128)
void gemm_bf16_small_kernel(const bf16* __restrict__ A, const bf16* __restrict__ Bt,
                            float* __restrict__ Cf, bf16* __restrict__ Cb,
                            const float* __restrict__ Res,
                            int M, int N, int K, int act, float alpha)
{
    const int lane = threadIdx.x & 31;
    const int wave = threadIdx.x >> 5;
    const int hlf  = lane >> 4;
    const int l15  = lane & 15;
    const int row0 = blockIdx.x * 64 + wave * 16;
    int arow = row0 + l15; if (arow > M - 1) arow = M - 1;
    const bf16* Arow = A + (size_t)arow * K;
    const int nbase = blockIdx.y * 64;

    v8f acc[4] = {};
    for (int k = 0; k < K; k += 32) {
        BQ af;
        af.q[0] = *(const uint4*)(Arow + k + hlf * 8);
        af.q[1] = *(const uint4*)(Arow + k + 16 + hlf * 8);
        if (k + 32 < K) __builtin_prefetch(Arow + k + 32, 0, 1);
#pragma unroll
        for (int nt = 0; nt < 4; ++nt) {
            int bn = nbase + nt * 16 + l15; if (bn > N - 1) bn = N - 1;
            const bf16* Brow = Bt + (size_t)bn * K + k + hlf * 8;
            BQ bq;
            bq.q[0] = *(const uint4*)(Brow);
            bq.q[1] = *(const uint4*)(Brow + 16);
            acc[nt] = __builtin_amdgcn_wmma_f32_16x16x32_bf16(
                false, af.v, false, bq.v, (short)0, acc[nt], false, false);
        }
    }
#pragma unroll
    for (int nt = 0; nt < 4; ++nt) {
        int col = nbase + nt * 16 + l15;
        if (col >= N) continue;
#pragma unroll
        for (int r = 0; r < 8; ++r) {
            int row = row0 + hlf * 8 + r;
            if (row >= M) continue;
            float v = acc[nt][r];
            if (act) v = siluf(v);
            if (Res) v = (v + Res[(size_t)row * N + col]) * alpha;
            else     v = v * alpha;
            if (Cf) Cf[(size_t)row * N + col] = v;
            if (Cb) Cb[(size_t)row * N + col] = (bf16)v;
        }
    }
}

// ---------------- weight prep ----------------
__global__ void transpose_bt_kernel(const float* __restrict__ W, bf16* __restrict__ Bt,
                                    int K, int N, int Kp)
{
    int i = blockIdx.x * 256 + threadIdx.x;
    int total = N * Kp;
    if (i >= total) return;
    int n = i / Kp, k = i % Kp;
    Bt[i] = (k < K) ? (bf16)W[(size_t)k * N + n] : (bf16)0.0f;
}

// Bt[u][(i*H + h)] = W[h][i][u]   (for einsum 'eih,hiu->eu' as GEMM)
__global__ void permute_bil_kernel(const float* __restrict__ W, bf16* __restrict__ Bt,
                                   int H, int NI, int NU)
{
    int i = blockIdx.x * 256 + threadIdx.x;
    int total = NU * NI * H;
    if (i >= total) return;
    int u = i / (NI * H);
    int rem = i % (NI * H);
    int ii = rem / H, h = rem % H;
    Bt[i] = (bf16)W[((size_t)h * NI + ii) * NU + u];
}

// f32 -> bf16 with optional zero-padded widening (Cin -> Cout)
__global__ void convert_pad_kernel(const float* __restrict__ in, bf16* __restrict__ out,
                                   int Cin, int Cout, size_t total)
{
    size_t i = (size_t)blockIdx.x * 256 + threadIdx.x;
    if (i >= total) return;
    size_t r = i / Cout; int c = (int)(i % Cout);
    out[i] = (c < Cin) ? (bf16)in[r * Cin + c] : (bf16)0.0f;
}

// ---------------- elementwise / gather / scatter ----------------
__global__ void ew_mul_bf_kernel(const float* __restrict__ a, const float* __restrict__ b,
                                 bf16* __restrict__ o, size_t n)
{
    size_t i = (size_t)blockIdx.x * 256 + threadIdx.x;
    if (i >= n) return;
    o[i] = (bf16)(a[i] * b[i]);
}

__global__ void gather_mul_kernel(const float* __restrict__ a, const float* __restrict__ b,
                                  const int* __restrict__ gid, float* __restrict__ o,
                                  int W, size_t n)
{
    size_t i = (size_t)blockIdx.x * 256 + threadIdx.x;
    if (i >= n) return;
    size_t j = i / W; int c = (int)(i % W);
    o[i] = a[(size_t)gid[j] * W + c] * b[i];
}

__global__ void swap_add_kernel(const float* __restrict__ u1, const float* __restrict__ u2,
                                const int* __restrict__ sw, float* __restrict__ o,
                                float alpha, int W, size_t n)
{
    size_t i = (size_t)blockIdx.x * 256 + threadIdx.x;
    if (i >= n) return;
    size_t e = i / W; int c = (int)(i % W);
    o[i] = (u1[i] + u2[(size_t)sw[e] * W + c]) * alpha;
}

// x = (skip + x4 + (u1 + u2[id_swap])*c2) * c3 ; f32 + bf16 out
__global__ void merge_kernel(const float* __restrict__ skip, const float* __restrict__ x4,
                             const float* __restrict__ u1, const float* __restrict__ u2,
                             const int* __restrict__ sw, float* __restrict__ of,
                             bf16* __restrict__ ob, int W, size_t n)
{
    size_t i = (size_t)blockIdx.x * 256 + threadIdx.x;
    if (i >= n) return;
    size_t e = i / W; int c = (int)(i % W);
    float x3 = (u1[i] + u2[(size_t)sw[e] * W + c]) * INV_SQRT2f;
    float v = (skip[i] + x3 + x4[i]) * INV_SQRT3f;
    of[i] = v; ob[i] = (bf16)v;
}

__global__ void addscale_kernel(const float* __restrict__ a, const float* __restrict__ b,
                                float* __restrict__ of, bf16* __restrict__ ob,
                                float alpha, size_t n)
{
    size_t i = (size_t)blockIdx.x * 256 + threadIdx.x;
    if (i >= n) return;
    float v = (a[i] + b[i]) * alpha;
    of[i] = v;
    if (ob) ob[i] = (bf16)v;
}

__global__ void zero_kernel(float* __restrict__ p, size_t n)
{
    size_t i = (size_t)blockIdx.x * 256 + threadIdx.x;
    if (i < n) p[i] = 0.0f;
}

// segment_sum: acc[idx[e], c] += a[e,c]*b[e,c]
__global__ void scatter_add_kernel(const float* __restrict__ a, const float* __restrict__ b,
                                   const int* __restrict__ idx, float* __restrict__ acc,
                                   int W, size_t n)
{
    size_t i = (size_t)blockIdx.x * 256 + threadIdx.x;
    if (i >= n) return;
    size_t e = i / W; int c = (int)(i % W);
    atomicAdd(&acc[(size_t)idx[e] * W + c], a[i] * b[i]);
}

// cat[e] = [ h[idx_s[e]](128) | h[idx_t[e]](128) | m[e](512) ]  (bf16, width 768)
__global__ void build_cat_kernel(const bf16* __restrict__ hb, const bf16* __restrict__ mb,
                                 const int* __restrict__ is_, const int* __restrict__ it_,
                                 bf16* __restrict__ cat, size_t n)
{
    size_t i = (size_t)blockIdx.x * 256 + threadIdx.x;
    if (i >= n) return;
    size_t e = i / 768; int c = (int)(i % 768);
    bf16 v;
    if (c < 128)      v = hb[(size_t)is_[e] * 128 + c];
    else if (c < 256) v = hb[(size_t)it_[e] * 128 + (c - 128)];
    else              v = mb[(size_t)e * 512 + (c - 256)];
    cat[i] = v;
}

// ============================================================================
// Per-edge bilinear stage-1/2:
//   m2[k,h]   = src[gid[e*K+k], h]
//   sumk[n,h] = sum_k sph[e,k,n] * m2[k,h]
//   r[i,h]    = sum_n rbfW1[e,i,n] * sumk[n,h]      -> rout[e, i*H + h] (bf16)
// ============================================================================
template<int K, int NS, int NSP, int NI, int H>
__global__ __launch_bounds__(256)
void bilin_kernel(const float* __restrict__ sph, const float* __restrict__ rbfW1,
                  const float* __restrict__ src, const int* __restrict__ gid,
                  bf16* __restrict__ rout)
{
    __shared__ float s_m[K][H];
    __shared__ float s_sph[K][NSP];
    __shared__ float s_sum[NSP][H];
    __shared__ float s_w[NI][NSP];
    const int e = blockIdx.x;
    const int tid = threadIdx.x;
    for (int idx = tid; idx < K * H; idx += 256) {
        int kk = idx / H, c = idx % H;
        int srow = gid[(size_t)e * K + kk];
        s_m[kk][c] = src[(size_t)srow * H + c];
    }
    for (int idx = tid; idx < K * NS; idx += 256) {
        int kk = idx / NS, nn = idx % NS;
        s_sph[kk][nn] = sph[(size_t)e * K * NS + idx];
    }
    for (int idx = tid; idx < NI * NS; idx += 256) {
        int ii = idx / NS, nn = idx % NS;
        s_w[ii][nn] = rbfW1[(size_t)e * NI * NS + idx];
    }
    __syncthreads();
    for (int idx = tid; idx < NS * H; idx += 256) {
        int nn = idx / H, c = idx % H;
        float s = 0.f;
#pragma unroll
        for (int kk = 0; kk < K; ++kk) s += s_sph[kk][nn] * s_m[kk][c];
        s_sum[nn][c] = s;
    }
    __syncthreads();
    for (int idx = tid; idx < NI * H; idx += 256) {
        int ii = idx / H, c = idx % H;
        float s = 0.f;
#pragma unroll
        for (int nn = 0; nn < NS; ++nn) s += s_w[ii][nn] * s_sum[nn][c];
        rout[(size_t)e * NI * H + idx] = (bf16)s;
    }
}

// ============================================================================
extern "C" void kernel_launch(void* const* d_in, const int* in_sizes, int n_in,
                              void* d_out, int out_size, void* d_ws, size_t ws_size,
                              hipStream_t stream)
{
    (void)in_sizes; (void)n_in; (void)out_size; (void)ws_size;
    const int E = 50000, D = 512, NA = 5000, EA = 128;
    const int NI4 = 25000;                    // N_INTERM
    const size_t ED = (size_t)E * D;

    // ---- inputs (setup_inputs() order) ----
    const float* h        = (const float*)d_in[0];
    const float* m        = (const float*)d_in[1];
    const float* rbf4     = (const float*)d_in[2];
    const float* cbf4     = (const float*)d_in[3];
    const float* rbfW1_4  = (const float*)d_in[4];
    const float* sph4     = (const float*)d_in[5];
    const float* rbf3     = (const float*)d_in[6];
    const float* rbfW1_3  = (const float*)d_in[7];
    const float* sph3     = (const float*)d_in[8];
    const float* rbf_h    = (const float*)d_in[9];
    const float* W_dense  = (const float*)d_in[10];
    const float* Wq_db    = (const float*)d_in[11];
    const float* Wq_rbf   = (const float*)d_in[12];
    const float* Wq_cbf   = (const float*)d_in[13];
    const float* Wq_bil   = (const float*)d_in[14];
    const float* Wq_down  = (const float*)d_in[15];
    const float* Wq_upca  = (const float*)d_in[16];
    const float* Wq_upac  = (const float*)d_in[17];
    const float* Wt_ba    = (const float*)d_in[18];
    const float* Wt_rbf   = (const float*)d_in[19];
    const float* Wt_bil   = (const float*)d_in[20];
    const float* Wt_down  = (const float*)d_in[21];
    const float* Wt_upca  = (const float*)d_in[22];
    const float* Wt_upac  = (const float*)d_in[23];
    const float* W_before = (const float*)d_in[24];
    const float* W_after  = (const float*)d_in[25];
    const float* W_au_rbf = (const float*)d_in[26];
    const float* W_au_den = (const float*)d_in[27];
    const float* W_au_res = (const float*)d_in[28];
    const float* W_concat = (const float*)d_in[29];
    const float* W_resm   = (const float*)d_in[30];
    const int* id_swap    = (const int*)d_in[32];
    const int* id4_intm   = (const int*)d_in[34];
    const int* id4_abd    = (const int*)d_in[35];
    const int* id3_ba     = (const int*)d_in[37];
    const int* idx_s      = (const int*)d_in[39];
    const int* idx_t      = (const int*)d_in[40];

    float* out_h = (float*)d_out;                       // (5000,128)
    float* out_m = (float*)d_out + (size_t)NA * EA;     // (50000,512)

    // ---- workspace bump allocator ----
    char* base = (char*)d_ws; size_t off = 0;
    auto alloc = [&](size_t bytes) -> void* {
        off = (off + 255) & ~(size_t)255;
        void* p = base + off; off += bytes; return p;
    };
    float* skip = (float*)alloc(ED * 4);
    float* t1   = (float*)alloc(ED * 4);
    float* t2   = (float*)alloc(ED * 4);
    float* x4b  = (float*)alloc(ED * 4);
    float* xb   = (float*)alloc(ED * 4);
    float* mn   = (float*)alloc(ED * 4);
    bf16* mbf   = (bf16*)alloc(ED * 2);   // m (bf16), later reused for m_new (bf16)
    bf16* bbf0  = (bf16*)alloc(ED * 2);
    bf16* bbf1  = (bf16*)alloc(ED * 2);
    bf16* catbf = (bf16*)alloc((size_t)E * 768 * 2);
    bf16* R     = (bf16*)alloc((size_t)E * 2048 * 2);   // shared by quad (2048) & trip (1024)
    bf16* rbf4b = (bf16*)alloc((size_t)E * 32 * 2);
    bf16* rbf3b = (bf16*)alloc((size_t)E * 32 * 2);
    bf16* rbfhb = (bf16*)alloc((size_t)E * 32 * 2);
    bf16* cbf4b = (bf16*)alloc((size_t)NI4 * 32 * 2);
    float* sq0  = (float*)alloc((size_t)E * 64 * 4);    // xq2 / xt2
    float* sq1  = (float*)alloc((size_t)NI4 * 64 * 4);  // cbf proj
    float* sq2  = (float*)alloc((size_t)NI4 * 64 * 4);  // gathered*cbf
    float* sq3  = (float*)alloc((size_t)E * 32 * 4);    // quad bilinear out
    bf16*  sq3b = (bf16*)alloc((size_t)E * 32 * 2);
    float* sq4  = (float*)alloc((size_t)E * 64 * 4);    // trip bilinear out
    bf16*  sq4b = (bf16*)alloc((size_t)E * 64 * 2);
    float* aacc = (float*)alloc((size_t)NA * D * 4);    // segment sum accumulator
    bf16*  aaccb= (bf16*)alloc((size_t)NA * D * 2);
    float* xa   = (float*)alloc((size_t)NA * EA * 4);
    bf16*  xab  = (bf16*)alloc((size_t)NA * EA * 2);
    bf16*  yab  = (bf16*)alloc((size_t)NA * EA * 2);
    bf16*  hnb  = (bf16*)alloc((size_t)NA * EA * 2);
    // transposed bf16 weights (Bt: N x Kpad)
    bf16* wdense_t = (bf16*)alloc((size_t)512 * 512 * 2);
    bf16* wqdb_t   = (bf16*)alloc((size_t)512 * 512 * 2);
    bf16* wqrbf_t  = (bf16*)alloc((size_t)512 * 32 * 2);
    bf16* wqcbf_t  = (bf16*)alloc((size_t)64 * 32 * 2);
    bf16* wqdown_t = (bf16*)alloc((size_t)64 * 512 * 2);
    bf16* wqupca_t = (bf16*)alloc((size_t)512 * 32 * 2);
    bf16* wqupac_t = (bf16*)alloc((size_t)512 * 32 * 2);
    bf16* wtba_t   = (bf16*)alloc((size_t)512 * 512 * 2);
    bf16* wtrbf_t  = (bf16*)alloc((size_t)512 * 32 * 2);
    bf16* wtdown_t = (bf16*)alloc((size_t)64 * 512 * 2);
    bf16* wtupca_t = (bf16*)alloc((size_t)512 * 64 * 2);
    bf16* wtupac_t = (bf16*)alloc((size_t)512 * 64 * 2);
    bf16* wb_t[2];  for (int i = 0; i < 2; ++i) wb_t[i] = (bf16*)alloc((size_t)512 * 512 * 2);
    bf16* wa_t[4];  for (int i = 0; i < 4; ++i) wa_t[i] = (bf16*)alloc((size_t)512 * 512 * 2);
    bf16* waurbf_t = (bf16*)alloc((size_t)512 * 32 * 2);
    bf16* wauden_t = (bf16*)alloc((size_t)128 * 512 * 2);
    bf16* waures_t[6]; for (int i = 0; i < 6; ++i) waures_t[i] = (bf16*)alloc((size_t)128 * 128 * 2);
    bf16* wcat_t   = (bf16*)alloc((size_t)512 * 768 * 2);
    bf16* wr_t[2];  for (int i = 0; i < 2; ++i) wr_t[i] = (bf16*)alloc((size_t)512 * 512 * 2);
    bf16* qbil_t   = (bf16*)alloc((size_t)32 * 2048 * 2);
    bf16* tbil_t   = (bf16*)alloc((size_t)64 * 1024 * 2);

    // ---- launch helpers ----
    auto gemm = [&](const bf16* A, const bf16* Bt, float* Cf, bf16* Cb, const float* Res,
                    int M, int N, int K, int act, float alpha) {
        if (N >= 128) {
            dim3 g((unsigned)((M + 127) / 128), (unsigned)((N + 127) / 128));
            gemm_bf16_big_kernel<<<g, 128, 0, stream>>>(A, Bt, Cf, Cb, Res, M, N, K, act, alpha);
        } else {
            dim3 g((unsigned)((M + 63) / 64), (unsigned)((N + 63) / 64));
            gemm_bf16_small_kernel<<<g, 128, 0, stream>>>(A, Bt, Cf, Cb, Res, M, N, K, act, alpha);
        }
    };
    auto tbt = [&](const float* W, bf16* Bt, int K, int N, int Kp) {
        int tot = N * Kp;
        transpose_bt_kernel<<<(tot + 255) / 256, 256, 0, stream>>>(W, Bt, K, N, Kp);
    };
    auto cpad = [&](const float* in, bf16* o, size_t rows, int Cin, int Cout) {
        size_t tot = rows * Cout;
        convert_pad_kernel<<<(unsigned)((tot + 255) / 256), 256, 0, stream>>>(in, o, Cin, Cout, tot);
    };
    auto fgrid = [](size_t n) { return dim3((unsigned)((n + 255) / 256)); };
    // residual layer: x = (x + silu(silu(x@W0)@W1)) / sqrt(2), in place on (xf, xbf)
    auto resid = [&](float* xf, bf16* xbf, const bf16* W0t, const bf16* W1t,
                     bf16* ybf, int M, int W) {
        gemm(xbf, W0t, nullptr, ybf, nullptr, M, W, W, 1, 1.0f);
        gemm(ybf, W1t, xf, xbf, xf, M, W, W, 1, INV_SQRT2f);
    };

    // ---- weight prep ----
    tbt(W_dense, wdense_t, 512, 512, 512);
    tbt(Wq_db,   wqdb_t,   512, 512, 512);
    tbt(Wq_rbf,  wqrbf_t,  16, 512, 32);
    tbt(Wq_cbf,  wqcbf_t,  16, 64, 32);
    tbt(Wq_down, wqdown_t, 512, 64, 512);
    tbt(Wq_upca, wqupca_t, 32, 512, 32);
    tbt(Wq_upac, wqupac_t, 32, 512, 32);
    tbt(Wt_ba,   wtba_t,   512, 512, 512);
    tbt(Wt_rbf,  wtrbf_t,  16, 512, 32);
    tbt(Wt_down, wtdown_t, 512, 64, 512);
    tbt(Wt_upca, wtupca_t, 64, 512, 64);
    tbt(Wt_upac, wtupac_t, 64, 512, 64);
    for (int i = 0; i < 2; ++i) tbt(W_before + (size_t)i * 512 * 512, wb_t[i], 512, 512, 512);
    for (int i = 0; i < 4; ++i) tbt(W_after  + (size_t)i * 512 * 512, wa_t[i], 512, 512, 512);
    tbt(W_au_rbf, waurbf_t, 16, 512, 32);
    tbt(W_au_den, wauden_t, 512, 128, 512);
    for (int i = 0; i < 6; ++i) tbt(W_au_res + (size_t)i * 128 * 128, waures_t[i], 128, 128, 128);
    tbt(W_concat, wcat_t, 768, 512, 768);
    for (int i = 0; i < 2; ++i) tbt(W_resm + (size_t)i * 512 * 512, wr_t[i], 512, 512, 512);
    {   // bilinear weights: Bt[u][(i*H+h)] = W[h,i,u]
        int tq = 32 * 32 * 64, tt = 64 * 16 * 64;
        permute_bil_kernel<<<(tq + 255) / 256, 256, 0, stream>>>(Wq_bil, qbil_t, 64, 32, 32);
        permute_bil_kernel<<<(tt + 255) / 256, 256, 0, stream>>>(Wt_bil, tbil_t, 64, 16, 64);
    }
    // ---- activation conversions ----
    cpad(m,     mbf,   E, 512, 512);
    cpad(rbf4,  rbf4b, E, 16, 32);
    cpad(rbf3,  rbf3b, E, 16, 32);
    cpad(rbf_h, rbfhb, E, 16, 32);
    cpad(cbf4,  cbf4b, NI4, 16, 32);

    // ---- skip connection ----
    gemm(mbf, wdense_t, skip, nullptr, nullptr, E, 512, 512, 1, 1.0f);   // silu(m@W_dense_ca)

    // ---- quadruplet branch ----
    gemm(mbf, wqdb_t, t1, nullptr, nullptr, E, 512, 512, 1, 1.0f);       // silu(m@Wq_db)
    gemm(rbf4b, wqrbf_t, t2, nullptr, nullptr, E, 512, 32, 0, 1.0f);     // rbf4@Wq_rbf
    ew_mul_bf_kernel<<<fgrid(ED), 256, 0, stream>>>(t1, t2, bbf0, ED);   // hadamard -> bf16
    gemm(bbf0, wqdown_t, sq0, nullptr, nullptr, E, 64, 512, 1, 1.0f);    // silu(.@Wq_down)
    gemm(cbf4b, wqcbf_t, sq1, nullptr, nullptr, NI4, 64, 32, 0, 1.0f);   // cbf4@Wq_cbf
    gather_mul_kernel<<<fgrid((size_t)NI4 * 64), 256, 0, stream>>>(
        sq0, sq1, id4_intm, sq2, 64, (size_t)NI4 * 64);                  // gather * cbf
    bilin_kernel<16, 16, 16, 32, 64><<<E, 256, 0, stream>>>(sph4, rbfW1_4, sq2, id4_abd, R);
    gemm(R, qbil_t, sq3, sq3b, nullptr, E, 32, 2048, 0, 1.0f);           // 'eih,hiu->eu'
    gemm(sq3b, wqupca_t, t1, nullptr, nullptr, E, 512, 32, 1, 1.0f);
    gemm(sq3b, wqupac_t, t2, nullptr, nullptr, E, 512, 32, 1, 1.0f);
    swap_add_kernel<<<fgrid(ED), 256, 0, stream>>>(t1, t2, id_swap, x4b, INV_SQRT2f, 512, ED);

    // ---- triplet branch ----
    gemm(mbf, wtba_t, t1, nullptr, nullptr, E, 512, 512, 1, 1.0f);
    gemm(rbf3b, wtrbf_t, t2, nullptr, nullptr, E, 512, 32, 0, 1.0f);
    ew_mul_bf_kernel<<<fgrid(ED), 256, 0, stream>>>(t1, t2, bbf1, ED);
    gemm(bbf1, wtdown_t, sq0, nullptr, nullptr, E, 64, 512, 1, 1.0f);    // xt2 (E,64)
    bilin_kernel<16, 7, 8, 16, 64><<<E, 256, 0, stream>>>(sph3, rbfW1_3, sq0, id3_ba, R);
    gemm(R, tbil_t, sq4, sq4b, nullptr, E, 64, 1024, 0, 1.0f);
    gemm(sq4b, wtupca_t, t1, nullptr, nullptr, E, 512, 64, 1, 1.0f);
    gemm(sq4b, wtupac_t, t2, nullptr, nullptr, E, 512, 64, 1, 1.0f);

    // ---- merge + residual stacks ----
    merge_kernel<<<fgrid(ED), 256, 0, stream>>>(skip, x4b, t1, t2, id_swap, xb, bbf0, 512, ED);
    resid(xb, bbf0, wb_t[0], wb_t[1], bbf1, E, 512);                     // W_before (1 layer)
    addscale_kernel<<<fgrid(ED), 256, 0, stream>>>(m, xb, mn, mbf, INV_SQRT2f, ED); // m_new
    resid(mn, mbf, wa_t[0], wa_t[1], bbf1, E, 512);                      // W_after layer 0
    resid(mn, mbf, wa_t[2], wa_t[3], bbf1, E, 512);                      // W_after layer 1

    // ---- atom update ----
    gemm(rbfhb, waurbf_t, t1, nullptr, nullptr, E, 512, 32, 0, 1.0f);    // rbf_h@W_au_rbf
    zero_kernel<<<fgrid((size_t)NA * D), 256, 0, stream>>>(aacc, (size_t)NA * D);
    scatter_add_kernel<<<fgrid(ED), 256, 0, stream>>>(mn, t1, idx_t, aacc, 512, ED);
    cpad(aacc, aaccb, NA, 512, 512);
    gemm(aaccb, wauden_t, xa, xab, nullptr, NA, 128, 512, 1, 1.0f);      // silu(.@W_au_dense)
    for (int l = 0; l < 3; ++l)
        resid(xa, xab, waures_t[2 * l], waures_t[2 * l + 1], yab, NA, 128);
    addscale_kernel<<<fgrid((size_t)NA * EA), 256, 0, stream>>>(
        h, xa, out_h, hnb, INV_SQRT2f, (size_t)NA * EA);                 // h_out

    // ---- concat / edge re-embedding ----
    build_cat_kernel<<<fgrid((size_t)E * 768), 256, 0, stream>>>(
        hnb, mbf, idx_s, idx_t, catbf, (size_t)E * 768);
    gemm(catbf, wcat_t, t2, bbf1, nullptr, E, 512, 768, 1, 1.0f);        // silu(cat@W_concat)
    resid(t2, bbf1, wr_t[0], wr_t[1], bbf0, E, 512);                     // W_resm (1 layer)
    addscale_kernel<<<fgrid(ED), 256, 0, stream>>>(mn, t2, out_m, nullptr, INV_SQRT2f, ED); // m_out
}
